// FocusMapGenerator_44839458571005
// MI455X (gfx1250) — matrix operand order repaired and checked by reading, not compiled, not run on hardware
//
#include <hip/hip_runtime.h>
#include <math.h>

typedef __attribute__((ext_vector_type(16))) _Float16 v16h;
typedef __attribute__((ext_vector_type(8)))  _Float16 v8h;
typedef __attribute__((ext_vector_type(8)))  float    v8f;

#define DI __device__ __forceinline__

// ======================= reduction helper =======================
DI float blk_reduce_sum(float v, float* sh) {
  int t = threadIdx.x;
  sh[t] = v; __syncthreads();
  for (int s = blockDim.x >> 1; s > 0; s >>= 1) {
    if (t < s) sh[t] += sh[t + s];
    __syncthreads();
  }
  float r = sh[0]; __syncthreads();
  return r;
}

// ======================= WMMA GEMM =======================
// A:  [M,K] f16 row-major (activations, NHWC / im2col)
// Bt: [N,K] f16 row-major (weights, torch (Cout, Cin*kh*kw) layout)
// C:  [M,N] f32 row-major
// One wave computes a 16x64 strip: 1 A fragment reused over 4 B fragments.
__global__ __launch_bounds__(256) void k_wmma_gemm(
    const _Float16* __restrict__ A, const _Float16* __restrict__ Bt,
    float* __restrict__ Cm, int M, int N, int K)
{
  const int wib  = threadIdx.x >> 5;
  const int lane = threadIdx.x & 31;
  const int gw   = blockIdx.x * 8 + wib;
  const int nW   = N >> 6;               // 64-wide wave columns
  const int tm   = gw / nW;
  const int tn   = gw % nW;
  if (tm >= (M >> 4)) return;            // wave-uniform guard (EXEC stays all-ones)
  const int half = lane >> 4;
  const int l15  = lane & 15;

  const _Float16* arow = A  + (size_t)(tm * 16 + l15) * K + half * 16;
  const _Float16* brow = Bt + (size_t)(tn * 64 + l15) * K + half * 16;

  v8f acc0 = {}, acc1 = {}, acc2 = {}, acc3 = {};
  for (int k0 = 0; k0 < K; k0 += 32) {
    __builtin_prefetch(arow + k0 + 64, 0, 1);
    // A fragment: lanes 0-15 hold K=k0..k0+15; lanes 16-31 hold K=k0+16..k0+31
    v8h alo = *(const v8h*)(arow + k0);
    v8h ahi = *(const v8h*)(arow + k0 + 8);
    v16h a = __builtin_shufflevector(alo, ahi, 0,1,2,3,4,5,6,7,8,9,10,11,12,13,14,15);
    // B fragments: lane half selects 16-K block, contiguous along K of column n
    v16h b0 = *(const v16h*)(brow + k0);
    v16h b1 = *(const v16h*)(brow + (size_t)16 * K + k0);
    v16h b2 = *(const v16h*)(brow + (size_t)32 * K + k0);
    v16h b3 = *(const v16h*)(brow + (size_t)48 * K + k0);
    acc0 = __builtin_amdgcn_wmma_f32_16x16x32_f16(false, a, false, b0, (short)0, acc0, false, false);
    acc1 = __builtin_amdgcn_wmma_f32_16x16x32_f16(false, a, false, b1, (short)0, acc1, false, false);
    acc2 = __builtin_amdgcn_wmma_f32_16x16x32_f16(false, a, false, b2, (short)0, acc2, false, false);
    acc3 = __builtin_amdgcn_wmma_f32_16x16x32_f16(false, a, false, b3, (short)0, acc3, false, false);
  }
  // C/D layout: VGPR r -> row tm*16 + half*8 + r, col = tn*64 + (lane&15) + 16*tile
  float* crow = Cm + (size_t)(tm * 16 + half * 8) * N + tn * 64 + l15;
#pragma unroll
  for (int r = 0; r < 8; ++r) {
    size_t o = (size_t)r * N;
    crow[o + 0]  = acc0[r];
    crow[o + 16] = acc1[r];
    crow[o + 32] = acc2[r];
    crow[o + 48] = acc3[r];
  }
}

// ======================= pack / cast kernels =======================
// NCHW fp32 -> [M=S*HW, K=C] f16 (i.e. NHWC f16)
__global__ void k_pack_nchw_to_mk(const float* __restrict__ X, _Float16* __restrict__ A,
                                  int C, int HW, long total) {
  long i = (long)blockIdx.x * blockDim.x + threadIdx.x;
  if (i >= total) return;
  int  c  = (int)(i % C);
  long m  = i / C;
  long s  = m / HW;
  int  hw = (int)(m % HW);
  A[i] = (_Float16)X[(s * C + c) * (long)HW + hw];
}

// same, with per-(sample,channel) scale (SE module fused into pack)
__global__ void k_pack_nchw_to_mk_scaled(const float* __restrict__ X, const float* __restrict__ S,
                                         _Float16* __restrict__ A, int C, int HW, long total) {
  long i = (long)blockIdx.x * blockDim.x + threadIdx.x;
  if (i >= total) return;
  int  c  = (int)(i % C);
  long m  = i / C;
  long s  = m / HW;
  int  hw = (int)(m % HW);
  A[i] = (_Float16)(X[(s * C + c) * (long)HW + hw] * S[s * C + c]);
}

__global__ void k_cast_f32_to_f16(const float* __restrict__ W, _Float16* __restrict__ Wh, long n) {
  long i = (long)blockIdx.x * blockDim.x + threadIdx.x;
  if (i < n) Wh[i] = (_Float16)W[i];
}

// im2col for 3x3 pad1: NCHW fp32 -> [M=S*H*W, K=C*9] f16
__global__ void k_im2col3_f16(const float* __restrict__ X, _Float16* __restrict__ A,
                              int C, int Hh, int Ww, long total) {
  long i = (long)blockIdx.x * blockDim.x + threadIdx.x;
  if (i >= total) return;
  int  K  = C * 9;
  int  k  = (int)(i % K);
  long m  = i / K;
  int  c  = k / 9;  int r = k % 9; int ky = r / 3, kx = r % 3;
  int  HW = Hh * Ww;
  long s  = m / HW; int hw = (int)(m % HW);
  int  h  = hw / Ww, w = hw % Ww;
  int  y  = h + ky - 1, x = w + kx - 1;
  float v = 0.f;
  if ((unsigned)y < (unsigned)Hh && (unsigned)x < (unsigned)Ww)
    v = X[(s * C + c) * (long)HW + y * Ww + x];
  A[i] = (_Float16)v;
}

// ======================= unpack kernels ([M,N] -> NCHW) =======================
__global__ void k_unpack_bias(const float* __restrict__ G, const float* __restrict__ bias,
                              float* __restrict__ Y, int N, int HW, long total) {
  long i = (long)blockIdx.x * blockDim.x + threadIdx.x;
  if (i >= total) return;
  int  c  = (int)(i % N);
  long m  = i / N;
  long s  = m / HW; int hw = (int)(m % HW);
  Y[(s * N + c) * (long)HW + hw] = G[i] + bias[c];
}

// X += relu(G + bias)   (attention residual)
__global__ void k_unpack_res_relu(const float* __restrict__ G, const float* __restrict__ bias,
                                  float* __restrict__ X, int N, int HW, long total) {
  long i = (long)blockIdx.x * blockDim.x + threadIdx.x;
  if (i >= total) return;
  int  c  = (int)(i % N);
  long m  = i / N;
  long s  = m / HW; int hw = (int)(m % HW);
  float v = G[i] + bias[c];
  v = v > 0.f ? v : 0.f;
  X[(s * N + c) * (long)HW + hw] += v;
}

// ======================= norm kernels =======================
// per-sample mean/var over a contiguous block (works for [M,N] grouped by sample and NCHW)
__global__ void k_gn_stats(const float* __restrict__ X, float* __restrict__ mv, long per) {
  __shared__ float sh[256];
  int s = blockIdx.x;
  const float* base = X + (size_t)s * per;
  float sum = 0.f, sq = 0.f;
  for (long i = threadIdx.x; i < per; i += blockDim.x) {
    float v = base[i]; sum += v; sq += v * v;
  }
  sum = blk_reduce_sum(sum, sh);
  sq  = blk_reduce_sum(sq, sh);
  if (threadIdx.x == 0) {
    float m = sum / (float)per;
    mv[s * 2] = m;
    mv[s * 2 + 1] = sq / (float)per - m * m;
  }
}

// GN apply on [M,N] -> NCHW with ReLU6 (FFN h1)
__global__ void k_gn_apply_mn_relu6(const float* __restrict__ G, const float* __restrict__ mv,
                                    const float* __restrict__ g, const float* __restrict__ b,
                                    float* __restrict__ Y, int N, int HW, long total) {
  long i = (long)blockIdx.x * blockDim.x + threadIdx.x;
  if (i >= total) return;
  int  c  = (int)(i % N);
  long m  = i / N;
  long s  = m / HW; int hw = (int)(m % HW);
  float rs = rsqrtf(mv[s * 2 + 1] + 1e-5f);
  float v  = (G[i] - mv[s * 2]) * rs * g[c] + b[c];
  v = fminf(fmaxf(v, 0.f), 6.f);
  Y[(s * N + c) * (long)HW + hw] = v;
}

// X = relu(X + GN(G))   (FFN output + block relu)
__global__ void k_gn_apply_mn_res_relu(const float* __restrict__ G, const float* __restrict__ mv,
                                       const float* __restrict__ g, const float* __restrict__ b,
                                       float* __restrict__ X, int N, int HW, long total) {
  long i = (long)blockIdx.x * blockDim.x + threadIdx.x;
  if (i >= total) return;
  int  c  = (int)(i % N);
  long m  = i / N;
  long s  = m / HW; int hw = (int)(m % HW);
  float rs = rsqrtf(mv[s * 2 + 1] + 1e-5f);
  float v  = (G[i] - mv[s * 2]) * rs * g[c] + b[c];
  long o = (s * N + c) * (long)HW + hw;
  X[o] = fmaxf(X[o] + v, 0.f);
}

// GN apply in place on NCHW with ReLU6 (FFN h2)
__global__ void k_gn_apply_nchw_relu6(float* __restrict__ X, const float* __restrict__ mv,
                                      const float* __restrict__ g, const float* __restrict__ b,
                                      int C, int HW, long total) {
  long i = (long)blockIdx.x * blockDim.x + threadIdx.x;
  if (i >= total) return;
  int  c = (int)((i / HW) % C);
  long s = i / ((long)C * HW);
  float rs = rsqrtf(mv[s * 2 + 1] + 1e-5f);
  float v  = (X[i] - mv[s * 2]) * rs * g[c] + b[c];
  X[i] = fminf(fmaxf(v, 0.f), 6.f);
}

// BatchNorm training stats: per channel over (S,HW)
__global__ void k_bn_stats(const float* __restrict__ X, float* __restrict__ mv,
                           int S, int C, int HW) {
  __shared__ float sh[256];
  int c = blockIdx.x;
  long per = (long)S * HW;
  float sum = 0.f, sq = 0.f;
  for (long i = threadIdx.x; i < per; i += blockDim.x) {
    long s = i / HW; int hw = (int)(i % HW);
    float v = X[(s * C + c) * (long)HW + hw];
    sum += v; sq += v * v;
  }
  sum = blk_reduce_sum(sum, sh);
  sq  = blk_reduce_sum(sq, sh);
  if (threadIdx.x == 0) {
    float m = sum / (float)per;
    mv[c * 2] = m;
    mv[c * 2 + 1] = sq / (float)per - m * m;
  }
}

__global__ void k_bn_apply_relu(float* __restrict__ X, const float* __restrict__ mv,
                                const float* __restrict__ g, const float* __restrict__ b,
                                int C, int HW, long total) {
  long i = (long)blockIdx.x * blockDim.x + threadIdx.x;
  if (i >= total) return;
  int c = (int)((i / HW) % C);
  float rs = rsqrtf(mv[c * 2 + 1] + 1e-5f);
  X[i] = fmaxf((X[i] - mv[c * 2]) * rs * g[c] + b[c], 0.f);
}

// ======================= direct convolutions (fp32) =======================
// act: 0 = none, 2 = sigmoid
__global__ void k_conv2d(const float* __restrict__ X, const float* __restrict__ W,
                         float* __restrict__ Y, int S, int Cin, int Hi, int Wi,
                         int Cout, int Kh, int Kw, int stride, int pad,
                         int Ho, int Wo, int act, long total) {
  long i = (long)blockIdx.x * blockDim.x + threadIdx.x;
  if (i >= total) return;
  int  wo = (int)(i % Wo); long t = i / Wo;
  int  ho = (int)(t % Ho); t /= Ho;
  int  co = (int)(t % Cout);
  int  s  = (int)(t / Cout);
  float acc = 0.f;
  for (int ci = 0; ci < Cin; ++ci) {
    const float* xb = X + (size_t)(s * Cin + ci) * Hi * Wi;
    const float* wb = W + ((size_t)co * Cin + ci) * Kh * Kw;
    for (int ky = 0; ky < Kh; ++ky) {
      int y = ho * stride - pad + ky;
      if ((unsigned)y >= (unsigned)Hi) continue;
      for (int kx = 0; kx < Kw; ++kx) {
        int x = wo * stride - pad + kx;
        if ((unsigned)x >= (unsigned)Wi) continue;
        acc += xb[(size_t)y * Wi + x] * wb[ky * Kw + kx];
      }
    }
  }
  if (act == 2) acc = 1.f / (1.f + expf(-acc));
  Y[i] = acc;
}

// ConvTranspose2d k=4, stride=2, pad=1. W: (Cin, Cout, 4, 4)
__global__ void k_convT2d(const float* __restrict__ X, const float* __restrict__ W,
                          float* __restrict__ Y, int S, int Cin, int Hi, int Wi,
                          int Cout, int Ho, int Wo, long total) {
  long i = (long)blockIdx.x * blockDim.x + threadIdx.x;
  if (i >= total) return;
  int  wo = (int)(i % Wo); long t = i / Wo;
  int  ho = (int)(t % Ho); t /= Ho;
  int  co = (int)(t % Cout);
  int  s  = (int)(t / Cout);
  float acc = 0.f;
  for (int ci = 0; ci < Cin; ++ci) {
    const float* xb = X + (size_t)(s * Cin + ci) * Hi * Wi;
    const float* wb = W + ((size_t)ci * Cout + co) * 16;
    for (int ky = 0; ky < 4; ++ky) {
      int ty = ho + 1 - ky;
      if (ty & 1) continue;
      int y = ty >> 1;
      if ((unsigned)y >= (unsigned)Hi) continue;
      for (int kx = 0; kx < 4; ++kx) {
        int tx = wo + 1 - kx;
        if (tx & 1) continue;
        int x = tx >> 1;
        if ((unsigned)x >= (unsigned)Wi) continue;
        acc += xb[(size_t)y * Wi + x] * wb[ky * 4 + kx];
      }
    }
  }
  Y[i] = acc;
}

// depthwise 3x3 pad1, W: (C,1,3,3)
__global__ void k_dwconv3(const float* __restrict__ X, const float* __restrict__ W,
                          float* __restrict__ Y, int C, int Hh, int Ww, long total) {
  long i = (long)blockIdx.x * blockDim.x + threadIdx.x;
  if (i >= total) return;
  int  x = (int)(i % Ww); long t = i / Ww;
  int  y = (int)(t % Hh); t /= Hh;
  int  c = (int)(t % C);
  const float* xb = X + (i - (size_t)x - (size_t)y * Ww); // base of this (s,c) plane
  const float* wb = W + (size_t)c * 9;
  float acc = 0.f;
  for (int ky = 0; ky < 3; ++ky) {
    int yy = y + ky - 1;
    if ((unsigned)yy >= (unsigned)Hh) continue;
    for (int kx = 0; kx < 3; ++kx) {
      int xx = x + kx - 1;
      if ((unsigned)xx >= (unsigned)Ww) continue;
      acc += xb[(size_t)yy * Ww + xx] * wb[ky * 3 + kx];
    }
  }
  Y[i] = acc;
}

// ======================= attention kernels =======================
DI long attn_off(int bi, int tok, int d, int cb, int ph, int oh) {
  int ow = oh, pw = ph;
  int ohw = oh * ow;
  int t  = tok / ohw; int r = tok - t * ohw;
  int oi = r / ow;    int oj = r - oi * ow;
  int pp = ph * pw;   int c = d / pp; int r2 = d - c * pp;
  int py = r2 / pw;   int px = r2 - py * pw;
  return ((long)((bi * 2 + t) * 256 + cb + c) * 48 + (oi * ph + py)) * 48 + (oj * pw + px);
}

// one block (128 thr) per (bi, n, m): Sc[bi][n][m] = <q_n, k_m> * scale
__global__ void k_attn_scores(const float* __restrict__ Q, const float* __restrict__ Kt,
                              float* __restrict__ Sc, int cb, int ph, int oh,
                              int Nt, float scale) {
  __shared__ float sh[128];
  int idx = blockIdx.x;
  int m  = idx % Nt; idx /= Nt;
  int n  = idx % Nt;
  int bi = idx / Nt;
  int D  = 64 * ph * ph;
  float sum = 0.f;
  for (int d = threadIdx.x; d < D; d += blockDim.x)
    sum += Q[attn_off(bi, n, d, cb, ph, oh)] * Kt[attn_off(bi, m, d, cb, ph, oh)];
  sum = blk_reduce_sum(sum, sh);
  if (threadIdx.x == 0) Sc[((long)bi * Nt + n) * Nt + m] = sum * scale;
}

// softmax over rows of length Nt (<=128), one block of 128 per row
__global__ void k_softmax_rows(float* __restrict__ Sc, int Nt) {
  __shared__ float sh[128];
  float* row = Sc + (size_t)blockIdx.x * Nt;
  int t = threadIdx.x;
  float v = (t < Nt) ? row[t] : -3.0e38f;
  sh[t] = v; __syncthreads();
  for (int s = 64; s > 0; s >>= 1) { if (t < s) sh[t] = fmaxf(sh[t], sh[t + s]); __syncthreads(); }
  float mx = sh[0]; __syncthreads();
  float e = (t < Nt) ? expf(v - mx) : 0.f;
  sh[t] = e; __syncthreads();
  for (int s = 64; s > 0; s >>= 1) { if (t < s) sh[t] += sh[t + s]; __syncthreads(); }
  float sum = sh[0];
  if (t < Nt) row[t] = e / sum;
}

// O[attn_off(bi,n,d)] = sum_m attn[bi][n][m] * V[attn_off(bi,m,d)]
__global__ void k_attn_out(const float* __restrict__ Sc, const float* __restrict__ V,
                           float* __restrict__ O, int cb, int ph, int oh,
                           int Nt, long total) {
  long i = (long)blockIdx.x * blockDim.x + threadIdx.x;
  if (i >= total) return;
  int D  = 64 * ph * ph;
  int d  = (int)(i % D); long r = i / D;
  int n  = (int)(r % Nt);
  int bi = (int)(r / Nt);
  const float* arow = Sc + ((long)bi * Nt + n) * Nt;
  float acc = 0.f;
  for (int m = 0; m < Nt; ++m)
    acc += arow[m] * V[attn_off(bi, m, d, cb, ph, oh)];
  O[attn_off(bi, n, d, cb, ph, oh)] = acc;
}

// ======================= SE kernels =======================
__global__ void k_hw_mean(const float* __restrict__ X, float* __restrict__ M, int HW) {
  __shared__ float sh[128];
  int sc = blockIdx.x;
  const float* base = X + (size_t)sc * HW;
  float sum = 0.f;
  for (int i = threadIdx.x; i < HW; i += blockDim.x) sum += base[i];
  sum = blk_reduce_sum(sum, sh);
  if (threadIdx.x == 0) M[sc] = sum / (float)HW;
}

// Out[s][j] = act( sum_k In[s][k] * W[j][k] );  act: 1 relu, 2 sigmoid
__global__ void k_fc(const float* __restrict__ In, const float* __restrict__ W,
                     float* __restrict__ Out, int Ki, int Ko, int act, int total) {
  int i = blockIdx.x * blockDim.x + threadIdx.x;
  if (i >= total) return;
  int j = i % Ko, s = i / Ko;
  float acc = 0.f;
  for (int k = 0; k < Ki; ++k) acc += In[s * Ki + k] * W[j * Ki + k];
  if (act == 1) acc = fmaxf(acc, 0.f);
  else if (act == 2) acc = 1.f / (1.f + expf(-acc));
  Out[i] = acc;
}

// ======================= layout utility kernels =======================
// imgA/imgB (4,3,192,192) -> X0 (8,3,192,192), sample = b*2 + t
__global__ void k_stack_imgs(const float* __restrict__ A, const float* __restrict__ Bm,
                             float* __restrict__ X, long per, long total) {
  long i = (long)blockIdx.x * blockDim.x + threadIdx.x;
  if (i >= total) return;
  long s = i / per, r = i % per;
  long b = s >> 1;
  X[i] = (s & 1) ? Bm[b * per + r] : A[b * per + r];
}

// X (8,256,2304) -> Y (4,512,2304): channel-concat of the two timesteps
__global__ void k_concat_pairs(const float* __restrict__ X, float* __restrict__ Y, long total) {
  long i = (long)blockIdx.x * blockDim.x + threadIdx.x;
  if (i >= total) return;
  int  hw = (int)(i % 2304); long r = i / 2304;
  int  c  = (int)(r % 512);
  long b  = r / 512;
  int  t  = c >> 8, cc = c & 255;
  Y[i] = X[((b * 2 + t) * 256 + cc) * 2304L + hw];
}

// ======================= host orchestration =======================
struct BlkP {
  const float *wq,*bq,*wk,*bk,*wv,*bv,*wo,*bo,*f1,*fg1,*fb1,*fdw,*fg2,*fb2,*sw1,*sw2,*f2,*fg3,*fb3;
};

extern "C" void kernel_launch(void* const* d_in, const int* in_sizes, int n_in,
                              void* d_out, int out_size, void* d_ws, size_t ws_size,
                              hipStream_t stream) {
  auto F = [&](int i) { return (const float*)d_in[i]; };

  // ---- parameter table (handles insertion-order or sorted-key flattening) ----
  const float *imgA = F(0), *imgB = F(1);
  const float *ew1,*eg1,*eb1,*ew2,*eg2,*eb2,*ew3,*eg3,*eb3;
  const float *dw1,*dg1,*db1,*dw2,*dg2,*db2,*dw3;
  BlkP blk[8];
  bool sorted = (n_in > 2 && in_sizes[2] == 256);   // sorted: blocks[0].bk first; insertion: enc.w1 (9408)
  if (!sorted) {
    int i = 2;
    ew1=F(i++); eg1=F(i++); eb1=F(i++); ew2=F(i++); eg2=F(i++); eb2=F(i++); ew3=F(i++); eg3=F(i++); eb3=F(i++);
    for (int b = 0; b < 8; ++b) {
      BlkP& p = blk[b];
      p.wq=F(i++); p.bq=F(i++); p.wk=F(i++); p.bk=F(i++); p.wv=F(i++); p.bv=F(i++);
      p.wo=F(i++); p.bo=F(i++); p.f1=F(i++); p.fg1=F(i++); p.fb1=F(i++); p.fdw=F(i++);
      p.fg2=F(i++); p.fb2=F(i++); p.sw1=F(i++); p.sw2=F(i++); p.f2=F(i++); p.fg3=F(i++); p.fb3=F(i++);
    }
    dw1=F(i++); dg1=F(i++); db1=F(i++); dw2=F(i++); dg2=F(i++); db2=F(i++); dw3=F(i++);
  } else {
    int i = 2;  // 'blocks' < 'dec' < 'enc'
    for (int b = 0; b < 8; ++b) {
      BlkP& p = blk[b];
      p.bk=F(i++); p.bo=F(i++); p.bq=F(i++); p.bv=F(i++); p.f1=F(i++); p.f2=F(i++);
      p.fb1=F(i++); p.fb2=F(i++); p.fb3=F(i++); p.fdw=F(i++); p.fg1=F(i++); p.fg2=F(i++); p.fg3=F(i++);
      p.sw1=F(i++); p.sw2=F(i++); p.wk=F(i++); p.wo=F(i++); p.wq=F(i++); p.wv=F(i++);
    }
    db1=F(i++); db2=F(i++); dg1=F(i++); dg2=F(i++); dw1=F(i++); dw2=F(i++); dw3=F(i++);
    eb1=F(i++); eb2=F(i++); eb3=F(i++); eg1=F(i++); eg2=F(i++); eg3=F(i++); ew1=F(i++); ew2=F(i++); ew3=F(i++);
  }

  // ---- workspace layout (~260 MB, all 256B aligned) ----
  char* ws = (char*)d_ws;
  const size_t OFF_X   = 0;                          // 8*256*2304*4   = 18,874,368
  const size_t OFF_A   = OFF_X  + 18874368;          // f16 A / h2     = 84,934,656
  const size_t OFF_B   = OFF_A  + 84934656;          // QKV+attcat / h1= 75,497,472
  const size_t OFF_G   = OFF_B  + 75497472;          // GEMM out       = 75,497,472
  const size_t OFF_W16 = OFF_G  + 75497472;          // f16 weights    =  4 MB
  const size_t OFF_MV  = OFF_W16 + 4194304;          // stats
  const size_t OFF_SE  = OFF_MV + 65536;             // SE buffers
  const size_t OFF_SC  = OFF_SE + 131072;            // attention scores

  float*     X   = (float*)(ws + OFF_X);
  _Float16*  Af  = (_Float16*)(ws + OFF_A);
  float*     Bf  = (float*)(ws + OFF_B);
  float*     G   = (float*)(ws + OFF_G);
  _Float16*  Wh  = (_Float16*)(ws + OFF_W16);
  float*     MV  = (float*)(ws + OFF_MV);
  float*     SEb = (float*)(ws + OFF_SE);
  float*     SC  = (float*)(ws + OFF_SC);

  auto cdiv = [](long a, long b) { return (unsigned)((a + b - 1) / b); };
  long tot;

  // ================= encoder =================
  tot = 8L * 3 * 192 * 192;
  k_stack_imgs<<<cdiv(tot, 256), 256, 0, stream>>>(imgA, imgB, Bf, 3L * 192 * 192, tot);

  tot = 8L * 64 * 192 * 192;                                    // conv1 7x7 pad3
  k_conv2d<<<cdiv(tot, 128), 128, 0, stream>>>(Bf, ew1, G, 8, 3, 192, 192, 64, 7, 7, 1, 3, 192, 192, 0, tot);
  k_bn_stats<<<64, 256, 0, stream>>>(G, MV, 8, 64, 192 * 192);
  k_bn_apply_relu<<<cdiv(tot, 256), 256, 0, stream>>>(G, MV, eg1, eb1, 64, 192 * 192, tot);

  tot = 8L * 128 * 96 * 96;                                     // conv2 3x3 s2
  k_conv2d<<<cdiv(tot, 128), 128, 0, stream>>>(G, ew2, Bf, 8, 64, 192, 192, 128, 3, 3, 2, 1, 96, 96, 0, tot);
  k_bn_stats<<<128, 256, 0, stream>>>(Bf, MV, 8, 128, 96 * 96);
  k_bn_apply_relu<<<cdiv(tot, 256), 256, 0, stream>>>(Bf, MV, eg2, eb2, 128, 96 * 96, tot);

  tot = 8L * 256 * 48 * 48;                                     // conv3 3x3 s2 -> X
  k_conv2d<<<cdiv(tot, 128), 128, 0, stream>>>(Bf, ew3, X, 8, 128, 96, 96, 256, 3, 3, 2, 1, 48, 48, 0, tot);
  k_bn_stats<<<256, 256, 0, stream>>>(X, MV, 8, 256, 2304);
  k_bn_apply_relu<<<cdiv(tot, 256), 256, 0, stream>>>(X, MV, eg3, eb3, 256, 2304, tot);

  // ================= transformer blocks =================
  const long M = 18432;                 // 8 * 48 * 48
  float* Q  = Bf;
  float* Kb = Bf + 4718592L;
  float* V  = Bf + 2 * 4718592L;
  float* AT = Bf + 3 * 4718592L;

  for (int b = 0; b < 8; ++b) {
    const BlkP& p = blk[b];

    // ----- multi-scale attention -----
    long MK = M * 256;
    k_pack_nchw_to_mk<<<cdiv(MK, 256), 256, 0, stream>>>(X, Af, 256, 2304, MK);

    const float* wqkv[3] = { p.wq, p.wk, p.wv };
    const float* bqkv[3] = { p.bq, p.bk, p.bv };
    float* oqkv[3] = { Q, Kb, V };
    for (int j = 0; j < 3; ++j) {
      k_cast_f32_to_f16<<<cdiv(65536, 256), 256, 0, stream>>>(wqkv[j], Wh, 65536);
      k_wmma_gemm<<<576, 256, 0, stream>>>(Af, Wh, G, (int)M, 256, 256);
      k_unpack_bias<<<cdiv(MK, 256), 256, 0, stream>>>(G, bqkv[j], oqkv[j], 256, 2304, MK);
    }

    for (int sc = 0; sc < 4; ++sc) {
      int ph = 48 >> sc, oh = 1 << sc;
      int Nt = 2 * oh * oh;
      int cb = sc * 64;
      int D  = 64 * ph * ph;
      float scl = 1.f / sqrtf((float)D);
      k_attn_scores<<<4 * Nt * Nt, 128, 0, stream>>>(Q, Kb, SC, cb, ph, oh, Nt, scl);
      k_softmax_rows<<<4 * Nt, 128, 0, stream>>>(SC, Nt);
      long td = 4L * Nt * D;
      k_attn_out<<<cdiv(td, 256), 256, 0, stream>>>(SC, V, AT, cb, ph, oh, Nt, td);
    }

    // wo 3x3 as im2col GEMM, then X += relu(out + bo)
    long MK9 = M * 2304;
    k_im2col3_f16<<<cdiv(MK9, 256), 256, 0, stream>>>(AT, Af, 256, 48, 48, MK9);
    k_cast_f32_to_f16<<<cdiv(589824, 256), 256, 0, stream>>>(p.wo, Wh, 589824);
    k_wmma_gemm<<<576, 256, 0, stream>>>(Af, Wh, G, (int)M, 256, 2304);
    k_unpack_res_relu<<<cdiv(MK, 256), 256, 0, stream>>>(G, p.bo, X, 256, 2304, MK);

    // ----- LocalityFeedForward -----
    k_pack_nchw_to_mk<<<cdiv(MK, 256), 256, 0, stream>>>(X, Af, 256, 2304, MK);
    k_cast_f32_to_f16<<<cdiv(262144, 256), 256, 0, stream>>>(p.f1, Wh, 262144);
    k_wmma_gemm<<<2304, 256, 0, stream>>>(Af, Wh, G, (int)M, 1024, 256);
    k_gn_stats<<<8, 256, 0, stream>>>(G, MV, 2304L * 1024);
    float* H1 = Bf;
    long t1 = M * 1024;
    k_gn_apply_mn_relu6<<<cdiv(t1, 256), 256, 0, stream>>>(G, MV, p.fg1, p.fb1, H1, 1024, 2304, t1);

    float* H2 = (float*)(ws + OFF_A);                // reuse f16-A region as fp32 h2
    long t2 = 8L * 1024 * 2304;
    k_dwconv3<<<cdiv(t2, 256), 256, 0, stream>>>(H1, p.fdw, H2, 1024, 48, 48, t2);
    k_gn_stats<<<8, 256, 0, stream>>>(H2, MV, 1024L * 2304);
    k_gn_apply_nchw_relu6<<<cdiv(t2, 256), 256, 0, stream>>>(H2, MV, p.fg2, p.fb2, 1024, 2304, t2);

    // SE
    float* Smean = SEb;
    float* S1v   = SEb + 8192;
    float* S2v   = SEb + 8192 + 512;
    k_hw_mean<<<8 * 1024, 128, 0, stream>>>(H2, Smean, 2304);
    k_fc<<<cdiv(8 * 64, 64), 64, 0, stream>>>(Smean, p.sw1, S1v, 1024, 64, 1, 8 * 64);
    k_fc<<<cdiv(8 * 1024, 256), 256, 0, stream>>>(S1v, p.sw2, S2v, 64, 1024, 2, 8 * 1024);

    // f2 projection with SE scale fused into the pack; then X = relu(X + GN(out))
    _Float16* A2 = (_Float16*)(ws + OFF_G);
    float*    G2 = (float*)(ws + OFF_G + 37748736);
    k_pack_nchw_to_mk_scaled<<<cdiv(t1, 256), 256, 0, stream>>>(H2, S2v, A2, 1024, 2304, t1);
    k_cast_f32_to_f16<<<cdiv(262144, 256), 256, 0, stream>>>(p.f2, Wh, 262144);
    k_wmma_gemm<<<576, 256, 0, stream>>>(A2, Wh, G2, (int)M, 256, 1024);
    k_gn_stats<<<8, 256, 0, stream>>>(G2, MV, 2304L * 256);
    k_gn_apply_mn_res_relu<<<cdiv(MK, 256), 256, 0, stream>>>(G2, MV, p.fg3, p.fb3, X, 256, 2304, MK);
  }

  // ================= decoder =================
  tot = 4L * 512 * 2304;
  k_concat_pairs<<<cdiv(tot, 256), 256, 0, stream>>>(X, Bf, tot);

  tot = 4L * 128 * 96 * 96;                                     // convT1 512->128
  k_convT2d<<<cdiv(tot, 128), 128, 0, stream>>>(Bf, dw1, G, 4, 512, 48, 48, 128, 96, 96, tot);
  k_bn_stats<<<128, 256, 0, stream>>>(G, MV, 4, 128, 96 * 96);
  k_bn_apply_relu<<<cdiv(tot, 256), 256, 0, stream>>>(G, MV, dg1, db1, 128, 96 * 96, tot);

  tot = 4L * 64 * 192 * 192;                                    // convT2 128->64
  k_convT2d<<<cdiv(tot, 128), 128, 0, stream>>>(G, dw2, Bf, 4, 128, 96, 96, 64, 192, 192, tot);
  k_bn_stats<<<64, 256, 0, stream>>>(Bf, MV, 4, 64, 192 * 192);
  k_bn_apply_relu<<<cdiv(tot, 256), 256, 0, stream>>>(Bf, MV, dg2, db2, 64, 192 * 192, tot);

  tot = 4L * 1 * 192 * 192;                                     // conv3 7x7 -> sigmoid -> d_out
  k_conv2d<<<cdiv(tot, 128), 128, 0, stream>>>(Bf, dw3, (float*)d_out,
                                               4, 64, 192, 192, 1, 7, 7, 1, 3, 192, 192, 2, tot);
}
